// VAE_884763263442
// MI455X (gfx1250) — matrix operand order repaired and checked by reading, not compiled, not run on hardware
//
#include <hip/hip_runtime.h>
#include <math.h>

// ---------------------------------------------------------------------------
// VAE-LSTM (T=512, IN=1024, H=2048, 2 layers, batch=1) for gfx1250 (MI455X).
// Strategy: convert weights fp32->f16 packed into WMMA B-tiles (L2-resident),
// one persistent kernel with a global-atomic grid barrier walking the
// sequential recurrence; every GEMV chunk runs through v_wmma_f32_16x16x32_f16
// with the input vector in A-matrix row M=0 (rows striped per the CDNA5 ISA
// layout: lanes 0/16 carry M=0).
// ---------------------------------------------------------------------------

constexpr int kH       = 2048;
constexpr int kT       = 512;
constexpr int kIn      = 1024;
constexpr int kNChunk  = kH / 16;   // 128 output chunks of 16
constexpr int kBlocks  = 32;
constexpr int kThreads = 128;       // 4 wave32 per block -> 128 waves total
constexpr int kKTilesH  = kH / 32;  // 64
constexpr int kKTilesIn = kIn / 32; // 32

typedef __attribute__((ext_vector_type(16))) _Float16 v16h;
typedef __attribute__((ext_vector_type(8)))  float    v8f;

// ----------------------------- small helpers -------------------------------

__device__ __forceinline__ v8f zero8() {
  v8f z;
#pragma unroll
  for (int j = 0; j < 8; ++j) z[j] = 0.f;
  return z;
}

__device__ __forceinline__ v8f wmma_acc(v16h a, v16h b, v8f c) {
  // D = A(16x32,f16) x B(32x16,f16) + C(16x16,f32)
  return __builtin_amdgcn_wmma_f32_16x16x32_f16(
      /*neg_a=*/false, a, /*neg_b=*/false, b,
      /*c_mod=*/(short)0, c, /*reuse_a=*/false, /*reuse_b=*/false);
}

__device__ __forceinline__ float sigmoidf_(float x) {
  return 1.f / (1.f + expf(-x));
}
__device__ __forceinline__ float softplusf_(float x) {
  return (x > 20.f) ? x : log1pf(expf(x));
}

// Monotonic global barrier: each block adds 1; barrier #ep done when
// counter >= ep*kBlocks.  Counter zeroed by init kernel every launch.
__device__ __forceinline__ void gsync(unsigned* bar, unsigned& ep) {
  __syncthreads();
  ++ep;
  if (threadIdx.x == 0) {
    __threadfence();
    atomicAdd(bar, 1u);
    const unsigned target = ep * (unsigned)kBlocks;
    while (*((volatile unsigned*)bar) < target) {
      __builtin_amdgcn_s_sleep(1);
    }
  }
  __syncthreads();
  __threadfence();
}

// --------------------------- A-fragment builders ---------------------------
// CDNA5 16-bit A (16x32, MxK): lanes 0-15 hold M=lane, lanes 16-31 hold
// M=lane-16.  For lane group 0-15: VGPR0..3 = K 0..7 pairs, VGPR4..7 = K16..23;
// lane group 16-31 shifts K by +8.  We only populate row M=0 (lanes 0,16),
// or rows M=0,1 (lanes 0,1,16,17) for the 2-row latent GEMM.

__device__ __forceinline__ v16h afrag_f16(const _Float16* __restrict__ x,
                                          int k0, int lane) {
  v16h a;
#pragma unroll
  for (int j = 0; j < 16; ++j) a[j] = (_Float16)0.f;
  if ((lane & 15) == 0) {
    const _Float16* xp = x + k0 + ((lane >> 4) << 3);
#pragma unroll
    for (int j = 0; j < 8; ++j) { a[j] = xp[j]; a[8 + j] = xp[16 + j]; }
  }
  return a;
}

__device__ __forceinline__ v16h afrag_f32(const float* __restrict__ x,
                                          int k0, int lane) {
  v16h a;
#pragma unroll
  for (int j = 0; j < 16; ++j) a[j] = (_Float16)0.f;
  if ((lane & 15) == 0) {
    const float* xp = x + k0 + ((lane >> 4) << 3);
#pragma unroll
    for (int j = 0; j < 8; ++j) {
      a[j]     = (_Float16)xp[j];
      a[8 + j] = (_Float16)xp[16 + j];
    }
  }
  return a;
}

__device__ __forceinline__ v16h afrag_2row(const _Float16* __restrict__ r0,
                                           const _Float16* __restrict__ r1,
                                           int k0, int lane) {
  v16h a;
#pragma unroll
  for (int j = 0; j < 16; ++j) a[j] = (_Float16)0.f;
  const int m = lane & 15;
  if (m < 2) {
    const _Float16* xp = (m == 0 ? r0 : r1) + k0 + ((lane >> 4) << 3);
#pragma unroll
    for (int j = 0; j < 8; ++j) { a[j] = xp[j]; a[8 + j] = xp[16 + j]; }
  }
  return a;
}

// --------------------------- GEMV accumulators -----------------------------
// Packed weight layout (made by pack_tiles_kernel): matrix (Nout x K) stored
// as tiles t = nt*(K/32)+kt, each tile 32(K)x16(N) f16, 16 K-values
// contiguous per lane => lane l reads its v16h at tile_base + l*16 halves
// (B layout: N = lane%16, 16 halves = K in table order, K-half = lane/16).

template <bool XF32>
__device__ __forceinline__ void gates_acc(const _Float16* __restrict__ Wt,
                                          int ktiles, int w, const void* xv,
                                          v8f& ai, v8f& af, v8f& ag, v8f& ao,
                                          int lane) {
  const _Float16* bi = Wt + ((long)(0 * kNChunk + w) * ktiles) * 512 + lane * 16;
  const _Float16* bf = Wt + ((long)(1 * kNChunk + w) * ktiles) * 512 + lane * 16;
  const _Float16* bg = Wt + ((long)(2 * kNChunk + w) * ktiles) * 512 + lane * 16;
  const _Float16* bo = Wt + ((long)(3 * kNChunk + w) * ktiles) * 512 + lane * 16;
  for (int kt = 0; kt < ktiles; ++kt) {
    v16h a;
    if (XF32) a = afrag_f32((const float*)xv, kt * 32, lane);
    else      a = afrag_f16((const _Float16*)xv, kt * 32, lane);
    if (kt + 1 < ktiles) {
      __builtin_prefetch(bi + (long)(kt + 1) * 512, 0, 1);
    }
    v16h vbi = *(const v16h*)(bi + (long)kt * 512);
    v16h vbf = *(const v16h*)(bf + (long)kt * 512);
    v16h vbg = *(const v16h*)(bg + (long)kt * 512);
    v16h vbo = *(const v16h*)(bo + (long)kt * 512);
    ai = wmma_acc(a, vbi, ai);
    af = wmma_acc(a, vbf, af);
    ag = wmma_acc(a, vbg, ag);
    ao = wmma_acc(a, vbo, ao);
  }
}

__device__ __forceinline__ void gemv2row(const _Float16* __restrict__ Wt,
                                         int ktiles, int w,
                                         const _Float16* r0, const _Float16* r1,
                                         v8f& acc, int lane) {
  const _Float16* b = Wt + ((long)w * ktiles) * 512 + lane * 16;
  for (int kt = 0; kt < ktiles; ++kt) {
    v16h a = afrag_2row(r0, r1, kt * 32, lane);
    acc = wmma_acc(a, *(const v16h*)(b + (long)kt * 512), acc);
  }
}

__device__ __forceinline__ void gemv1row(const _Float16* __restrict__ Wt,
                                         int ktiles, int w,
                                         const _Float16* x, v8f& acc, int lane) {
  const _Float16* b = Wt + ((long)w * ktiles) * 512 + lane * 16;
  for (int kt = 0; kt < ktiles; ++kt) {
    v16h a = afrag_f16(x, kt * 32, lane);
    acc = wmma_acc(a, *(const v16h*)(b + (long)kt * 512), acc);
  }
}

// ------------------------------- LSTM cell ---------------------------------
// One wave owns 16 hidden units (chunk w); c-state lives in registers
// (lanes 0..15) for the whole sequence.

template <bool XF32>
__device__ __forceinline__ void lstm_step(
    const _Float16* __restrict__ Wih, int kx_tiles, const void* x,
    const _Float16* __restrict__ Whh, const _Float16* __restrict__ hprev,
    const float* __restrict__ bih, const float* __restrict__ bhh,
    _Float16* __restrict__ hout, _Float16* __restrict__ hout2,
    float& c, int w, int lane) {
  v8f ai = zero8(), af = zero8(), ag = zero8(), ao = zero8();
  gates_acc<XF32>(Wih, kx_tiles, w, x, ai, af, ag, ao, lane);
  gates_acc<false>(Whh, kKTilesH, w, (const void*)hprev, ai, af, ag, ao, lane);
  if (lane < 16) {
    const int idx = w * 16 + lane;
    const float gi = ai[0] + bih[idx]            + bhh[idx];
    const float gf = af[0] + bih[kH + idx]       + bhh[kH + idx];
    const float gg = ag[0] + bih[2 * kH + idx]   + bhh[2 * kH + idx];
    const float go = ao[0] + bih[3 * kH + idx]   + bhh[3 * kH + idx];
    const float cn = sigmoidf_(gf) * c + sigmoidf_(gi) * tanhf(gg);
    const float hn = sigmoidf_(go) * tanhf(cn);
    c = cn;
    const _Float16 hh = (_Float16)hn;
    hout[idx] = hh;
    if (hout2) hout2[idx] = hh;
  }
}

// ------------------------------ prep kernels -------------------------------

__global__ void pack_tiles_kernel(const float* __restrict__ W,
                                  _Float16* __restrict__ dst,
                                  int Nout, int K) {
  const long total  = (long)Nout * K;
  const int  ktiles = K >> 5;
  for (long i = (long)blockIdx.x * blockDim.x + threadIdx.x; i < total;
       i += (long)gridDim.x * blockDim.x) {
    const long tile = i >> 9;           // 512 halves per 32x16 tile
    const int  rem  = (int)(i & 511);
    const int  ln   = rem >> 4;         // B lane (0..31): N = ln&15, K half = ln>>4
    const int  s    = rem & 15;         // element within lane
    const int  nl   = ln & 15;          // N within tile (0..15)
    const int  kl   = (s < 8) ? ((ln >> 4) << 3) + s : 16 + ((ln >> 4) << 3) + (s - 8);  // K within tile, table order
    const long nt   = tile / ktiles;
    const long kt   = tile - nt * ktiles;
    const long n    = nt * 16 + nl;
    const long k    = kt * 32 + kl;
    dst[i] = (_Float16)W[n * (long)K + k];
  }
}

__global__ void init_state_kernel(_Float16* h0, _Float16* h1, _Float16* xd,
                                  unsigned* bar) {
  const int i = blockIdx.x * blockDim.x + threadIdx.x;
  if (i < 2 * kH) { h0[i] = (_Float16)0.f; h1[i] = (_Float16)0.f; xd[i] = (_Float16)0.f; }
  if (i == 0) *bar = 0u;
}

// --------------------------- persistent kernel -----------------------------

__global__ void __launch_bounds__(kThreads)
vae_lstm_persistent(
    const float* __restrict__ features, const float* __restrict__ eps,
    const float* __restrict__ bih0e, const float* __restrict__ bhh0e,
    const float* __restrict__ bih1e, const float* __restrict__ bhh1e,
    const float* __restrict__ bmu,  const float* __restrict__ bvar,
    const float* __restrict__ bih0d, const float* __restrict__ bhh0d,
    const float* __restrict__ bih1d, const float* __restrict__ bhh1d,
    const float* __restrict__ boutp,
    const _Float16* __restrict__ Wih0e, const _Float16* __restrict__ Whh0e,
    const _Float16* __restrict__ Wih1e, const _Float16* __restrict__ Whh1e,
    const _Float16* __restrict__ Wmu,  const _Float16* __restrict__ Wvar,
    const _Float16* __restrict__ Wih0d, const _Float16* __restrict__ Whh0d,
    const _Float16* __restrict__ Wih1d, const _Float16* __restrict__ Whh1d,
    const _Float16* __restrict__ Wout,
    _Float16* h0buf, _Float16* h1buf, _Float16* xdec,
    unsigned* bar, float* __restrict__ out) {
  const int lane = threadIdx.x & 31;
  const int w    = (int)((blockIdx.x * kThreads + threadIdx.x) >> 5); // 0..127
  unsigned ep = 0;
  float c0 = 0.f, c1 = 0.f;   // cell state: this wave's 16 units (lanes 0..15)

  // ------------------------------ encoder ---------------------------------
  for (int t = 0; t < kT; ++t) {
    const int rp = t & 1, wp = rp ^ 1;  // double-buffered h vectors
    lstm_step<true>(Wih0e, kKTilesIn, (const void*)(features + (long)t * kIn),
                    Whh0e, h0buf + rp * kH, bih0e, bhh0e,
                    h0buf + wp * kH, nullptr, c0, w, lane);
    gsync(bar, ep);
    lstm_step<false>(Wih1e, kKTilesH, (const void*)(h0buf + wp * kH),
                     Whh1e, h1buf + rp * kH, bih1e, bhh1e,
                     h1buf + wp * kH, nullptr, c1, w, lane);
    gsync(bar, ep);
  }
  // after 512 steps the final h-vectors sit in buffer slot 0.

  // ------------------------------- latent ---------------------------------
  {
    v8f am = zero8(), av = zero8();
    gemv2row(Wmu,  kKTilesH, w, h0buf, h1buf, am, lane);  // rows M=0,1
    gemv2row(Wvar, kKTilesH, w, h0buf, h1buf, av, lane);
    float z0 = 0.f, z1 = 0.f;
    if (lane < 16) {
      const int idx = w * 16 + lane;
      const float mu0 = am[0] + bmu[idx];
      const float mu1 = am[1] + bmu[idx];
      const float sp0 = softplusf_(av[0] + bvar[idx]);
      const float sp1 = softplusf_(av[1] + bvar[idx]);
      out[idx]            = mu0;            // h_mu row 0
      out[kH + idx]       = mu1;            // h_mu row 1
      out[2 * kH + idx]   = logf(sp0);      // h_log_variance row 0
      out[3 * kH + idx]   = logf(sp1);      // h_log_variance row 1
      z0 = mu0 + eps[idx]      * sqrtf(sp0);  // exp(0.5*log sp) == sqrt(sp)
      z1 = mu1 + eps[kH + idx] * sqrtf(sp1);
    }
    gsync(bar, ep);                       // all reads of h-bufs complete
    if (lane < 16) {
      const int idx = w * 16 + lane;
      h0buf[idx] = (_Float16)z0;          // decoder dh0 init = z[0]
      h1buf[idx] = (_Float16)z1;          // decoder dh1 init = z[1]
    }
    gsync(bar, ep);
  }
  // c0/c1 carry straight over: decoder dc0/dc1 init = c_last (in registers).

  // ------------------------------- decoder --------------------------------
  float* __restrict__ dec = out + 4 * kH;
  for (int t = 0; t < kT; ++t) {
    const int rp = t & 1, wp = rp ^ 1;
    // layer0: x = previous dh1 (zeros at t=0, via xdec buffer)
    lstm_step<false>(Wih0d, kKTilesH, (const void*)(xdec + rp * kH),
                     Whh0d, h0buf + rp * kH, bih0d, bhh0d,
                     h0buf + wp * kH, nullptr, c0, w, lane);
    gsync(bar, ep);
    // layer1: also feeds next step's x (xdec) with dh1
    lstm_step<false>(Wih1d, kKTilesH, (const void*)(h0buf + wp * kH),
                     Whh1d, h1buf + rp * kH, bih1d, bhh1d,
                     h1buf + wp * kH, xdec + wp * kH, c1, w, lane);
    gsync(bar, ep);
    // output projection (1024 outputs -> waves 0..63 == blocks 0..15)
    if (w < (kIn / 16)) {
      v8f ap = zero8();
      gemv1row(Wout, kKTilesH, w, h1buf + wp * kH, ap, lane);
      if (lane < 16) {
        const int j = w * 16 + lane;
        dec[(long)(kT - 1 - t) * kIn + j] = ap[0] + boutp[j];  // outs[::-1]
      }
    }
  }
}

// ------------------------------- host entry --------------------------------

extern "C" void kernel_launch(void* const* d_in, const int* in_sizes, int n_in,
                              void* d_out, int out_size, void* d_ws, size_t ws_size,
                              hipStream_t stream) {
  (void)in_sizes; (void)n_in; (void)out_size; (void)ws_size;

  const float* features = (const float*)d_in[0];
  const float* eps      = (const float*)d_in[1];
  const float* Wih0e_f  = (const float*)d_in[2];
  const float* Whh0e_f  = (const float*)d_in[3];
  const float* bih0e    = (const float*)d_in[4];
  const float* bhh0e    = (const float*)d_in[5];
  const float* Wih1e_f  = (const float*)d_in[6];
  const float* Whh1e_f  = (const float*)d_in[7];
  const float* bih1e    = (const float*)d_in[8];
  const float* bhh1e    = (const float*)d_in[9];
  const float* Wmu_f    = (const float*)d_in[10];
  const float* bmu      = (const float*)d_in[11];
  const float* Wvar_f   = (const float*)d_in[12];
  const float* bvar     = (const float*)d_in[13];
  const float* Wih0d_f  = (const float*)d_in[14];
  const float* Whh0d_f  = (const float*)d_in[15];
  const float* bih0d    = (const float*)d_in[16];
  const float* bhh0d    = (const float*)d_in[17];
  const float* Wih1d_f  = (const float*)d_in[18];
  const float* Whh1d_f  = (const float*)d_in[19];
  const float* bih1d    = (const float*)d_in[20];
  const float* bhh1d    = (const float*)d_in[21];
  const float* Wout_f   = (const float*)d_in[22];
  const float* boutp    = (const float*)d_in[23];
  float* out = (float*)d_out;

  // carve d_ws: packed f16 weights + state buffers + barrier
  size_t off = 0;
  auto take = [&](size_t halves) -> _Float16* {
    _Float16* p = (_Float16*)((char*)d_ws + off);
    off += halves * sizeof(_Float16);
    off  = (off + 255) & ~(size_t)255;
    return p;
  };
  _Float16* tWih0e = take((size_t)4 * kH * kIn);
  _Float16* tWhh0e = take((size_t)4 * kH * kH);
  _Float16* tWih1e = take((size_t)4 * kH * kH);
  _Float16* tWhh1e = take((size_t)4 * kH * kH);
  _Float16* tWmu   = take((size_t)kH * kH);
  _Float16* tWvar  = take((size_t)kH * kH);
  _Float16* tWih0d = take((size_t)4 * kH * kH);
  _Float16* tWhh0d = take((size_t)4 * kH * kH);
  _Float16* tWih1d = take((size_t)4 * kH * kH);
  _Float16* tWhh1d = take((size_t)4 * kH * kH);
  _Float16* tWout  = take((size_t)kIn * kH);
  _Float16* h0buf  = take(2 * kH);
  _Float16* h1buf  = take(2 * kH);
  _Float16* xdec   = take(2 * kH);
  unsigned* bar    = (unsigned*)((char*)d_ws + off);

  auto pack = [&](const float* W, _Float16* dst, int Nout, int K) {
    long total  = (long)Nout * K;
    int threads = 256;
    long blocks = (total + threads - 1) / threads;
    if (blocks > 2048) blocks = 2048;
    pack_tiles_kernel<<<dim3((unsigned)blocks), dim3(threads), 0, stream>>>(W, dst, Nout, K);
  };
  pack(Wih0e_f, tWih0e, 4 * kH, kIn);
  pack(Whh0e_f, tWhh0e, 4 * kH, kH);
  pack(Wih1e_f, tWih1e, 4 * kH, kH);
  pack(Whh1e_f, tWhh1e, 4 * kH, kH);
  pack(Wmu_f,   tWmu,   kH,     kH);
  pack(Wvar_f,  tWvar,  kH,     kH);
  pack(Wih0d_f, tWih0d, 4 * kH, kH);
  pack(Whh0d_f, tWhh0d, 4 * kH, kH);
  pack(Wih1d_f, tWih1d, 4 * kH, kH);
  pack(Whh1d_f, tWhh1d, 4 * kH, kH);
  pack(Wout_f,  tWout,  kIn,    kH);

  init_state_kernel<<<dim3(16), dim3(256), 0, stream>>>(h0buf, h1buf, xdec, bar);

  vae_lstm_persistent<<<dim3(kBlocks), dim3(kThreads), 0, stream>>>(
      features, eps,
      bih0e, bhh0e, bih1e, bhh1e, bmu, bvar,
      bih0d, bhh0d, bih1d, bhh1d, boutp,
      tWih0e, tWhh0e, tWih1e, tWhh1e, tWmu, tWvar,
      tWih0d, tWhh0d, tWih1d, tWhh1d, tWout,
      h0buf, h1buf, xdec, bar, out);
}
